// TrfEdgeNetRand_79645873537757
// MI455X (gfx1250) — matrix operand order repaired and checked by reading, not compiled, not run on hardware
//
#include <hip/hip_runtime.h>
#include <hip/hip_bf16.h>
#include <math.h>

typedef __attribute__((ext_vector_type(16))) __bf16 v16bf;
typedef __attribute__((ext_vector_type(8)))  __bf16 v8bf;
typedef __attribute__((ext_vector_type(4)))  __bf16 v4bf;
typedef __attribute__((ext_vector_type(8)))  float  v8f;

// ---------- monotone float <-> uint encoding for atomic segment-max ----------
__device__ __forceinline__ unsigned encF(float x) {
    int i = __float_as_int(x);
    return (i < 0) ? ~(unsigned)i : ((unsigned)i | 0x80000000u);
}
__device__ __forceinline__ float decF(unsigned u) {
    int i = (u & 0x80000000u) ? (int)(u & 0x7fffffffu) : (int)(~u);
    return __int_as_float(i);
}
#define ENC_NEG_INF 0x007fffffu  // encF(-inf)

// =====================================================================
// Activation pre-pass: Xb = bf16(X), row-major [N][K]. 4 elems/thread.
// =====================================================================
__global__ void conv_x_bf16(const float* __restrict__ X, __bf16* __restrict__ Xb,
                            int total4)
{
    int i = blockIdx.x * blockDim.x + threadIdx.x;
    if (i >= total4) return;
    float4 v = ((const float4*)X)[i];
    v4bf o; o[0] = (__bf16)v.x; o[1] = (__bf16)v.y; o[2] = (__bf16)v.z; o[3] = (__bf16)v.w;
    ((v4bf*)Xb)[i] = o;
}

// =====================================================================
// Weight pre-pass: Wt[n*K + k] = bf16(W[k*Cout + n]), zero-padded to
// CoutPad columns -> B operand is 16 contiguous bf16 per lane per
// K-step (k = 16*(lane/16)+i, same order as A), branchless b128s.
// =====================================================================
__global__ void conv_w_bf16t(const float* __restrict__ W, __bf16* __restrict__ Wt,
                             int K, int Cout, int CoutPad)
{
    int idx = blockIdx.x * blockDim.x + threadIdx.x;
    if (idx >= CoutPad * K) return;
    int n = idx / K;
    int k = idx - n * K;
    float v = (n < Cout) ? W[(size_t)k * Cout + n] : 0.f;
    Wt[idx] = (__bf16)v;
}

// =====================================================================
// Dense GEMM: Y[N x Cout] = Xb[N x K](bf16) @ Wt + bias (+acc) (+relu)
// One wave32 computes an 80x16 strip: 5 M-tiles sharing one B register
// block; inner loop is pure loads + 5x v_wmma_f32_16x16x32_bf16 (no cvt).
// N multiple of 80 (100000 = 1250*80). K multiple of 32.
// =====================================================================
#define MT 5
__global__ void gemm_bias_wmma(const __bf16* __restrict__ Xb,   // [N][K] bf16
                               const __bf16* __restrict__ Wt,   // [CoutPad][K] bf16
                               const float* __restrict__ bias,
                               const float* __restrict__ acc,
                               float* __restrict__ Y,
                               int N, int K, int Cout, int relu)
{
    const int lane = threadIdx.x & 31;
    const int wave = threadIdx.x >> 5;
    const int groupsM = N / (16 * MT);
    const int tilesN  = (Cout + 15) >> 4;
    int g = blockIdx.x * (blockDim.x >> 5) + wave;
    if (g >= groupsM * tilesN) return;          // wave-uniform -> EXEC all-1 for WMMA
    const int mGroup  = g % groupsM;
    const int colTile = g / groupsM;

    // A layout (16x32 bf16): lane&15 = M row; elements i <-> K 2*kb+i,
    // kb = (lane>>4)*8
    const int mA  = lane & 15;
    const int kbA = (lane >> 4) * 8;
    // B layout (32x16 bf16): lane&15 = N col; elements i <-> K kbB+i, kbB = (lane>>4)*16
    const int nB  = lane & 15;
    const int kbB = (lane >> 4) * 16;
    const int col = colTile * 16 + nB;          // always < CoutPad (padded)

    const __bf16* Abase = Xb + (size_t)(mGroup * 16 * MT + mA) * (size_t)K;
    const __bf16* Wcol  = Wt + (size_t)col * (size_t)K;

    v8f c[MT];
#pragma unroll
    for (int m = 0; m < MT; ++m) c[m] = (v8f){};

    for (int k0 = 0; k0 < K; k0 += 32) {
        v16bf b = *(const v16bf*)(Wcol + k0 + kbB);          // 32B contiguous
        v16bf a[MT];
#pragma unroll
        for (int m = 0; m < MT; ++m) {
            const __bf16* Arow = Abase + (size_t)m * 16 * (size_t)K;
            v8bf lo = *(const v8bf*)(Arow + k0 + 2 * kbA);        // K 2kb..2kb+7
            v8bf hi = *(const v8bf*)(Arow + k0 + 2 * kbA + 8);    // K 2kb+8..2kb+15
            a[m] = __builtin_shufflevector(lo, hi, 0, 1, 2, 3, 4, 5, 6, 7,
                                           8, 9, 10, 11, 12, 13, 14, 15);
        }
#pragma unroll
        for (int m = 0; m < MT; ++m)
            c[m] = __builtin_amdgcn_wmma_f32_16x16x32_bf16(false, a[m], false, b,
                                                           (short)0, c[m], false, false);
    }

    const bool colOK = (col < Cout);
    const float bcol = bias[colOK ? col : 0];   // clamped, branchless
    if (colOK) {
#pragma unroll
        for (int m = 0; m < MT; ++m) {
            const int rbase = mGroup * 16 * MT + m * 16 + (lane >> 4) * 8;
#pragma unroll
            for (int r = 0; r < 8; ++r) {
                size_t idx = (size_t)(rbase + r) * (size_t)Cout + col;
                float v = c[m][r] + bcol;
                if (acc)  v += acc[idx];
                if (relu) v = v > 0.f ? v : 0.f;
                Y[idx] = v;
            }
        }
    }
}

// =====================================================================
// Per-layer scratch init: agg = 0, m = enc(-inf), s = 0
// =====================================================================
__global__ void seg_init(unsigned* __restrict__ menc, float* __restrict__ ssum,
                         float* __restrict__ agg, int N, int C)
{
    int i = blockIdx.x * blockDim.x + threadIdx.x;
    if (i < N * C) agg[i] = 0.f;
    if (i < N) { menc[i] = ENC_NEG_INF; ssum[i] = 0.f; }
}

// =====================================================================
// Pass A: alpha[e] = <q[dst], k[src] + edge_attr[e]@We> / sqrt(C)
// one wave per edge; lane = feature (coalesced 128B gathers, L2-resident
// q/k). We staged in LDS; edge_attr broadcast via lane shuffles.
// =====================================================================
__global__ void edge_alpha(const int* __restrict__ src, const int* __restrict__ dst,
                           const float* __restrict__ ea, const float* __restrict__ We,
                           const float* __restrict__ q, const float* __restrict__ k,
                           float* __restrict__ alpha, unsigned* __restrict__ menc,
                           int E, int C, float invSqrtC)
{
    __shared__ float sWe[16 * 256];
    for (int t = threadIdx.x; t < 16 * C; t += blockDim.x) sWe[t] = We[t];
    __syncthreads();

    const int wave = threadIdx.x >> 5, lane = threadIdx.x & 31;
    int e = blockIdx.x * (blockDim.x >> 5) + wave;
    if (e >= E) return;
    const int si = src[e], di = dst[e];
    float eav = (lane < 16) ? ea[(size_t)e * 16 + lane] : 0.f;

    float partial = 0.f;
    for (int f = lane; f < C; f += 32) {
        float ev = 0.f;
#pragma unroll
        for (int j = 0; j < 16; ++j)
            ev += __shfl(eav, j, 32) * sWe[j * C + f];
        partial += q[(size_t)di * C + f] * (k[(size_t)si * C + f] + ev);
    }
#pragma unroll
    for (int off = 16; off > 0; off >>= 1)
        partial += __shfl_xor(partial, off, 32);

    if (lane == 0) {
        float a = partial * invSqrtC;
        alpha[e] = a;
        atomicMax(&menc[di], encF(a));           // segment max (global_atomic_max_u32)
    }
}

// =====================================================================
// Pass B: p[e] = exp(alpha[e] - m[dst]); s[dst] += p[e]
// =====================================================================
__global__ void edge_p(const int* __restrict__ dst, const float* __restrict__ alpha,
                       const unsigned* __restrict__ menc,
                       float* __restrict__ p, float* __restrict__ ssum, int E)
{
    int i = blockIdx.x * blockDim.x + threadIdx.x;
    if (i >= E) return;
    int d = dst[i];
    float pv = __expf(alpha[i] - decF(menc[d]));
    p[i] = pv;
    atomicAdd(&ssum[d], pv);
}

// =====================================================================
// Pass C: agg[dst] += (p/s) * (v[src] + edge_attr@We)   (e recomputed)
// =====================================================================
__global__ void edge_agg(const int* __restrict__ src, const int* __restrict__ dst,
                         const float* __restrict__ ea, const float* __restrict__ We,
                         const float* __restrict__ v, const float* __restrict__ p,
                         const float* __restrict__ ssum, float* __restrict__ agg,
                         int E, int C)
{
    __shared__ float sWe[16 * 256];
    for (int t = threadIdx.x; t < 16 * C; t += blockDim.x) sWe[t] = We[t];
    __syncthreads();

    const int wave = threadIdx.x >> 5, lane = threadIdx.x & 31;
    int e = blockIdx.x * (blockDim.x >> 5) + wave;
    if (e >= E) return;
    const int si = src[e], di = dst[e];
    const float a = p[e] / (ssum[di] + 1e-16f);
    float eav = (lane < 16) ? ea[(size_t)e * 16 + lane] : 0.f;

    for (int f = lane; f < C; f += 32) {
        float ev = 0.f;
#pragma unroll
        for (int j = 0; j < 16; ++j)
            ev += __shfl(eav, j, 32) * sWe[j * C + f];
        atomicAdd(&agg[(size_t)di * C + f], a * (v[(size_t)si * C + f] + ev));
    }
}

// =====================================================================
// Host-side orchestration
// =====================================================================
extern "C" void kernel_launch(void* const* d_in, const int* in_sizes, int n_in,
                              void* d_out, int out_size, void* d_ws, size_t ws_size,
                              hipStream_t stream)
{
    const float* x  = (const float*)d_in[0];
    const int*   ei = (const int*)d_in[1];
    const float* ea = (const float*)d_in[2];

    const int N = in_sizes[0] / 128;   // 100000
    const int E = in_sizes[1] / 2;     // 800000
    const int* srcI = ei;
    const int* dstI = ei + E;

    // workspace layout (floats)
    float*  ws    = (float*)d_ws;
    size_t  nc    = (size_t)N * 256;
    float*  qbuf  = ws;
    float*  kbuf  = qbuf + nc;
    float*  vbuf  = kbuf + nc;
    float*  agg   = vbuf + nc;
    float*  h1    = agg  + nc;
    float*  h2    = h1   + nc;
    float*  alpha = h2   + nc;
    float*  pbuf  = alpha + E;
    float*  ssum  = pbuf + E;
    unsigned* menc = (unsigned*)(ssum + N);
    __bf16* wtScr = (__bf16*)(menc + N);          // 256*256 bf16 = 128 KB
    __bf16* xbScr = wtScr + 256 * 256;            // N*256 bf16 = 51.2 MB

    auto gemm = [&](const __bf16* Xb, const float* W, const float* b,
                    const float* acc, float* Y, int K, int Cout, int relu) {
        int CoutPad = (Cout + 15) & ~15;
        conv_w_bf16t<<<(CoutPad * K + 255) / 256, 256, 0, stream>>>(W, wtScr, K, Cout, CoutPad);
        int groups = (N / (16 * MT)) * (CoutPad / 16);
        gemm_bias_wmma<<<(groups + 3) / 4, 128, 0, stream>>>(Xb, wtScr, b, acc, Y,
                                                             N, K, Cout, relu);
    };

    auto layer = [&](const float* hin, int din, int dout, int base,
                     float* hout, int relu) {
        const float* Wq = (const float*)d_in[base + 0];
        const float* bq = (const float*)d_in[base + 1];
        const float* Wk = (const float*)d_in[base + 2];
        const float* bk = (const float*)d_in[base + 3];
        const float* Wv = (const float*)d_in[base + 4];
        const float* bv = (const float*)d_in[base + 5];
        const float* We = (const float*)d_in[base + 6];
        const float* Ws = (const float*)d_in[base + 7];
        const float* bs = (const float*)d_in[base + 8];

        // one activation-conversion pass reused by q/k/v/skip GEMMs
        int total4 = (N * din) / 4;
        conv_x_bf16<<<(total4 + 255) / 256, 256, 0, stream>>>(hin, xbScr, total4);

        gemm(xbScr, Wq, bq, nullptr, qbuf, din, dout, 0);
        gemm(xbScr, Wk, bk, nullptr, kbuf, din, dout, 0);
        gemm(xbScr, Wv, bv, nullptr, vbuf, din, dout, 0);

        seg_init<<<(N * dout + 255) / 256, 256, 0, stream>>>(menc, ssum, agg, N, dout);

        float invSqrtC = 1.0f / sqrtf((float)dout);
        edge_alpha<<<(E + 7) / 8, 256, 0, stream>>>(srcI, dstI, ea, We, qbuf, kbuf,
                                                    alpha, menc, E, dout, invSqrtC);
        edge_p<<<(E + 255) / 256, 256, 0, stream>>>(dstI, alpha, menc, pbuf, ssum, E);
        edge_agg<<<(E + 7) / 8, 256, 0, stream>>>(srcI, dstI, ea, We, vbuf, pbuf,
                                                  ssum, agg, E, dout);
        // out = agg + hin @ Ws + bs  (+ReLU for layers 1,2)
        gemm(xbScr, Ws, bs, agg, hout, din, dout, relu);
    };

    layer(x,  128, 128,  3, h1, 1);   // layer 1 + relu
    layer(h1, 128, 128, 12, h2, 1);   // layer 2 + relu
    layer(h2, 128, 256, 21, h1, 0);   // layer 3 (no relu)

    // classifier: [N,256] @ [256,10] + bc -> d_out
    const float* Wc = (const float*)d_in[30];
    const float* bc = (const float*)d_in[31];
    int total4 = (N * 256) / 4;
    conv_x_bf16<<<(total4 + 255) / 256, 256, 0, stream>>>(h1, xbScr, total4);
    gemm(xbScr, Wc, bc, nullptr, (float*)d_out, 256, 10, 0);
}
